// ReciprocalAttention_30983894073598
// MI455X (gfx1250) — matrix operand order repaired and checked by reading, not compiled, not run on hardware
//
#include <hip/hip_runtime.h>

// ================= problem constants (from reference) =================
// B=2, T=2048, C=1024, H=16, D=64, R=4, D_std=60
constexpr int Bb    = 2;
constexpr int Tt    = 2048;
constexpr int Cc    = 1024;
constexpr int Hh    = 16;
constexpr int Dd    = 64;
constexpr int DSTD  = 60;
constexpr int MROWS = Bb * Tt;   // 4096
constexpr int F3C   = 3 * Cc;    // 3072

typedef __attribute__((ext_vector_type(16))) __bf16 v16bf;
typedef __attribute__((ext_vector_type(8)))  __bf16 v8bf;
typedef __attribute__((ext_vector_type(8)))  float  v8f;
typedef __attribute__((ext_vector_type(4)))  float  f32x4;

// ---------------- small helpers ----------------
static __device__ inline __bf16 f2bf(float f) { return (__bf16)f; }   // hw cvt
static __device__ inline float bf2f(__bf16 b) { return (float)b; }
static __device__ inline float fast_rcp(float x) { return __builtin_amdgcn_rcpf(x); }
static __device__ inline float sigmoidf_(float x) {
  return fast_rcp(1.0f + __expf(-x));
}
static __device__ inline v8f vzero8() {
  v8f z;
#pragma unroll
  for (int i = 0; i < 8; ++i) z[i] = 0.0f;
  return z;
}
static __device__ inline v16bf cat16(v8bf lo, v8bf hi) {
  return __builtin_shufflevector(lo, hi, 0,1,2,3,4,5,6,7,8,9,10,11,12,13,14,15);
}

// WMMA D = A(16x32 bf16) * B(32x16 bf16) + C(16x16 f32)
static __device__ inline v8f wmma_bf(v16bf a, v16bf b, v8f c) {
  return __builtin_amdgcn_wmma_f32_16x16x32_bf16(false, a, false, b, (short)0, c, false, false);
}

// ---- fragment loaders (per CDNA5 ISA 7.12.2 16-bit layouts, wave32) ----
// A 16x32: lane m = l&15, hi = l>>4.  K(e) = (e<8 ? hi*8+e : 16+hi*8+(e-8))
static __device__ inline v16bf load_a_bf(const __bf16* A, int lda, int m0, int k0, int lane) {
  int m = lane & 15, hi = lane >> 4;
  const __bf16* base = A + (size_t)(m0 + m) * lda + k0;
  v8bf lo = *(const v8bf*)(base + hi * 8);
  v8bf hv = *(const v8bf*)(base + 16 + hi * 8);
  return cat16(lo, hv);
}
// B 32x16: lane n = l&15, hi = l>>4.  K(e) = (e<8 ? hi*8+e : 16+hi*8+(e-8))
// Bm is row-major with rows indexed by n (columns of B) and cols by k.
static __device__ inline v16bf load_b_bf(const __bf16* Bm, int ldb, int n0, int k0, int lane) {
  int n = lane & 15, hi = lane >> 4;
  const __bf16* base = Bm + (size_t)(n0 + n) * ldb + k0;
  v8bf lo = *(const v8bf*)(base + hi * 8);
  v8bf hv = *(const v8bf*)(base + 16 + hi * 8);
  return cat16(lo, hv);
}

// =====================================================================
// Kernel 0: streaming fp32 -> bf16 conversion (one-time, BW-trivial).
// Each thread converts 8 elements: 2x b128 loads -> 1x b128 store.
// =====================================================================
__global__ __launch_bounds__(256) void cvt_bf16_kernel(
    const float* __restrict__ src, __bf16* __restrict__ dst, int n8)
{
  int i = blockIdx.x * 256 + threadIdx.x;
  if (i >= n8) return;
  const f32x4* p = (const f32x4*)src + 2 * (size_t)i;
  f32x4 a = p[0], b = p[1];
  v8bf r;
  r[0] = f2bf(a[0]); r[1] = f2bf(a[1]); r[2] = f2bf(a[2]); r[3] = f2bf(a[3]);
  r[4] = f2bf(b[0]); r[5] = f2bf(b[1]); r[6] = f2bf(b[2]); r[7] = f2bf(b[3]);
  *((v8bf*)dst + i) = r;
}

// =====================================================================
// Kernel 1: fused = x @ W_attn^T, scale Q/K per head-dim, relayout:
//   Qbf,Kbf : [B,H,T,D] bf16    Vt : [B,H,D,T] bf16 (transposed)
// One 16x16 tile per wave; 8 waves / block. Pure b128+WMMA inner loop.
// =====================================================================
__global__ __launch_bounds__(256) void qkv_gemm_kernel(
    const __bf16* __restrict__ xbf, const __bf16* __restrict__ Wabf,
    const float* __restrict__ w_std, const float* __restrict__ w_rec,
    const float* __restrict__ skip_std, const float* __restrict__ skip_low,
    __bf16* __restrict__ Qbf, __bf16* __restrict__ Kbf, __bf16* __restrict__ Vt)
{
  const int lane = threadIdx.x & 31, wave = threadIdx.x >> 5;
  const int tile = blockIdx.x * 8 + wave;
  const int tilesN = F3C / 16;                    // 192
  const int tm = tile / tilesN, tn = tile % tilesN;
  const int m0 = tm * 16, n0 = tn * 16;

  v8f acc = vzero8();
#pragma unroll 2
  for (int k0 = 0; k0 < Cc; k0 += 32) {
    v16bf a = load_a_bf(xbf,  Cc, m0, k0, lane);
    v16bf b = load_b_bf(Wabf, Cc, n0, k0, lane);
    acc = wmma_bf(a, b, acc);
  }

  // epilogue: column (feature) is per-lane constant in C-layout
  const int n = lane & 15, hi = lane >> 4;
  const int f = n0 + n;
  const int sect = f >> 10;            // 0=q 1=k 2=v
  const int w = f & (Cc - 1);
  const int h = w >> 6, d = w & 63;
  float scale = 1.0f;
  if (sect < 2) {
    float s = (d < DSTD) ? w_std[h] * sigmoidf_(skip_std[0])
                         : w_rec[h] * sigmoidf_(skip_low[0]);
    scale = sqrtf(fmaxf(s, 1e-8f));
  }
#pragma unroll
  for (int r = 0; r < 8; ++r) {
    int row = m0 + r + hi * 8;          // global (b,t)
    int bq = row >> 11, t = row & (Tt - 1);
    __bf16 v = f2bf(acc[r] * scale);
    size_t slab = (size_t)bq * Hh + h;
    if (sect == 0)      Qbf[(slab * Tt + t) * Dd + d] = v;
    else if (sect == 1) Kbf[(slab * Tt + t) * Dd + d] = v;
    else                Vt [(slab * Dd + d) * Tt + t] = v;
  }
}

// =====================================================================
// Kernel 2: causal flash attention, one 16-row q-tile per wave.
// kv chunk = 32: 4 score WMMAs + online softmax + 4 PV WMMAs.
// P converted C-layout -> A-layout through a private LDS tile.
// =====================================================================
__global__ __launch_bounds__(256) void attn_kernel(
    const __bf16* __restrict__ Qbf, const __bf16* __restrict__ Kbf,
    const __bf16* __restrict__ Vt,  const float* __restrict__ rwr_alpha,
    __bf16* __restrict__ ybf)
{
  __shared__ __attribute__((aligned(64))) __bf16 plds[8][16 * 32];
  const int lane = threadIdx.x & 31, wave = threadIdx.x >> 5;
  const int wid = blockIdx.x * 8 + wave;          // 0 .. B*H*T/16-1
  const int qt = wid & (Tt / 16 - 1);             // q tile
  const int bh = wid >> 7;                        // 0..31
  const int h = bh & (Hh - 1);
  const int q0 = qt * 16;
  const int n = lane & 15, hi = lane >> 4;

  const __bf16* Qs = Qbf + (size_t)bh * Tt * Dd;
  const __bf16* Ks = Kbf + (size_t)bh * Tt * Dd;
  const __bf16* Vs = Vt  + (size_t)bh * Dd * Tt;
  __bf16* myp = &plds[wave][0];

  // Q fragments for d-chunks 0 and 32 (reused across all kv chunks)
  v16bf aq0 = load_a_bf(Qs, Dd, q0, 0,  lane);
  v16bf aq1 = load_a_bf(Qs, Dd, q0, 32, lane);

  float rowmax[8], rowsum[8];
#pragma unroll
  for (int r = 0; r < 8; ++r) { rowmax[r] = -1e30f; rowsum[r] = 0.0f; }
  v8f acc0 = vzero8(), acc1 = vzero8(), acc2 = vzero8(), acc3 = vzero8();

  const float sm = 0.125f;                        // 1/sqrt(64)
  const int kv_end = q0 + 16;                     // causal bound (exclusive)
  for (int kv0 = 0; kv0 < kv_end; kv0 += 32) {
    // ---- scores: S0 = cols kv0..+15, S1 = cols kv0+16..+31
    v8f s0 = vzero8(), s1 = vzero8();
    s0 = wmma_bf(aq0, load_b_bf(Ks, Dd, kv0,      0,  lane), s0);
    s0 = wmma_bf(aq1, load_b_bf(Ks, Dd, kv0,      32, lane), s0);
    s1 = wmma_bf(aq0, load_b_bf(Ks, Dd, kv0 + 16, 0,  lane), s1);
    s1 = wmma_bf(aq1, load_b_bf(Ks, Dd, kv0 + 16, 32, lane), s1);

    const int kg0 = kv0 + n, kg1 = kv0 + 16 + n;  // per-lane global k cols
#pragma unroll
    for (int r = 0; r < 8; ++r) {
      int qg = q0 + r + hi * 8;
      float v0 = (kg0 <= qg) ? s0[r] * sm : -1e30f;
      float v1 = (kg1 <= qg) ? s1[r] * sm : -1e30f;
      // row max over the 16 lanes that own this row (lanes 0-15 / 16-31)
      float mx = fmaxf(v0, v1);
      mx = fmaxf(mx, __shfl_xor(mx, 1));
      mx = fmaxf(mx, __shfl_xor(mx, 2));
      mx = fmaxf(mx, __shfl_xor(mx, 4));
      mx = fmaxf(mx, __shfl_xor(mx, 8));
      float nm = fmaxf(rowmax[r], mx);
      float fr = __expf(rowmax[r] - nm);
      rowmax[r] = nm;
      float p0 = __expf(v0 - nm);
      float p1 = __expf(v1 - nm);
      rowsum[r] = rowsum[r] * fr + p0 + p1;       // per-lane partial sum
      acc0[r] *= fr; acc1[r] *= fr; acc2[r] *= fr; acc3[r] *= fr;
      // stage P (row-major 16x32) for the A-layout reload
      myp[(r + hi * 8) * 32 + n]      = f2bf(p0);
      myp[(r + hi * 8) * 32 + 16 + n] = f2bf(p1);
    }
    // P as A fragment (compiler inserts s_wait_dscnt for the RAW)
    v16bf ap = load_a_bf(myp, 32, 0, 0, lane);
    // ---- y += P * V ; V^T layout makes B fragments contiguous along t
    acc0 = wmma_bf(ap, load_b_bf(Vs, Tt, 0,  kv0, lane), acc0);
    acc1 = wmma_bf(ap, load_b_bf(Vs, Tt, 16, kv0, lane), acc1);
    acc2 = wmma_bf(ap, load_b_bf(Vs, Tt, 32, kv0, lane), acc2);
    acc3 = wmma_bf(ap, load_b_bf(Vs, Tt, 48, kv0, lane), acc3);
  }

  // ---- finalize: normalize, alpha-blend with V, write bf16 y [B,T,C]
  const float alpha = fminf(fmaxf(rwr_alpha[h], 0.0f), 0.5f);
  const int b = bh >> 4;
#pragma unroll
  for (int r = 0; r < 8; ++r) {
    float rs = rowsum[r];
    rs += __shfl_xor(rs, 1);
    rs += __shfl_xor(rs, 2);
    rs += __shfl_xor(rs, 4);
    rs += __shfl_xor(rs, 8);
    float inv = fast_rcp(rs);
    int t = q0 + r + hi * 8;
    __bf16* orow = ybf + ((size_t)b * Tt + t) * Cc + h * Dd;
    float y;
    y = acc0[r] * inv; y = (1.0f - alpha) * y + alpha * bf2f(Vs[(size_t)(0  + n) * Tt + t]); orow[0  + n] = f2bf(y);
    y = acc1[r] * inv; y = (1.0f - alpha) * y + alpha * bf2f(Vs[(size_t)(16 + n) * Tt + t]); orow[16 + n] = f2bf(y);
    y = acc2[r] * inv; y = (1.0f - alpha) * y + alpha * bf2f(Vs[(size_t)(32 + n) * Tt + t]); orow[32 + n] = f2bf(y);
    y = acc3[r] * inv; y = (1.0f - alpha) * y + alpha * bf2f(Vs[(size_t)(48 + n) * Tt + t]); orow[48 + n] = f2bf(y);
  }
}

// =====================================================================
// Kernel 3: out = y @ W_proj^T  (all-bf16 inputs, f32 out)
// =====================================================================
__global__ __launch_bounds__(256) void proj_gemm_kernel(
    const __bf16* __restrict__ ybf, const __bf16* __restrict__ Wpbf,
    float* __restrict__ out)
{
  const int lane = threadIdx.x & 31, wave = threadIdx.x >> 5;
  const int tile = blockIdx.x * 8 + wave;
  const int tilesN = Cc / 16;                     // 64
  const int tm = tile / tilesN, tn = tile % tilesN;
  const int m0 = tm * 16, n0 = tn * 16;

  v8f acc = vzero8();
#pragma unroll 2
  for (int k0 = 0; k0 < Cc; k0 += 32) {
    v16bf a = load_a_bf(ybf,  Cc, m0, k0, lane);
    v16bf b = load_b_bf(Wpbf, Cc, n0, k0, lane);
    acc = wmma_bf(a, b, acc);
  }
  const int n = lane & 15, hi = lane >> 4;
#pragma unroll
  for (int r = 0; r < 8; ++r)
    out[(size_t)(m0 + r + hi * 8) * Cc + n0 + n] = acc[r];
}

// =====================================================================
extern "C" void kernel_launch(void* const* d_in, const int* in_sizes, int n_in,
                              void* d_out, int out_size, void* d_ws, size_t ws_size,
                              hipStream_t stream) {
  const float* x         = (const float*)d_in[0];
  const float* W_attn    = (const float*)d_in[1];
  const float* W_proj    = (const float*)d_in[2];
  const float* w_std     = (const float*)d_in[3];
  const float* w_rec     = (const float*)d_in[4];
  const float* skip_std  = (const float*)d_in[5];
  const float* skip_low  = (const float*)d_in[6];
  const float* rwr_alpha = (const float*)d_in[7];
  float* out = (float*)d_out;

  // workspace layout (bf16 elements):
  //   Q,K,Vt,y: 4 * B*T*C   x: B*T*C   Wa: 3*C*C   Wp: C*C   => 48 MB total
  const size_t nelem = (size_t)Bb * Tt * Cc;       // 4,194,304
  __bf16* Qbf  = (__bf16*)d_ws;
  __bf16* Kbf  = Qbf  + nelem;
  __bf16* Vt   = Kbf  + nelem;
  __bf16* ybf  = Vt   + nelem;
  __bf16* xbf  = ybf  + nelem;
  __bf16* Wabf = xbf  + nelem;
  __bf16* Wpbf = Wabf + (size_t)F3C * Cc;

  // 0) one-time fp32 -> bf16 conversions (streaming, L2-resident afterwards)
  cvt_bf16_kernel<<<(int)(nelem / 8 / 256),              256, 0, stream>>>(x,      xbf,  (int)(nelem / 8));
  cvt_bf16_kernel<<<(int)((size_t)F3C * Cc / 8 / 256),   256, 0, stream>>>(W_attn, Wabf, (int)((size_t)F3C * Cc / 8));
  cvt_bf16_kernel<<<(int)((size_t)Cc * Cc / 8 / 256),    256, 0, stream>>>(W_proj, Wpbf, (int)((size_t)Cc * Cc / 8));

  // 1) QKV GEMM: (4096x3072x1024), 256x192 tiles, 8 waves/block
  qkv_gemm_kernel<<<(MROWS / 16) * (F3C / 16) / 8, 256, 0, stream>>>(
      xbf, Wabf, w_std, w_rec, skip_std, skip_low, Qbf, Kbf, Vt);

  // 2) attention: B*H*(T/16) = 4096 wave-tiles, 8 waves/block
  attn_kernel<<<(Bb * Hh * (Tt / 16)) / 8, 256, 0, stream>>>(
      Qbf, Kbf, Vt, rwr_alpha, ybf);

  // 3) projection GEMM: (4096x1024x1024), 256x64 tiles
  proj_gemm_kernel<<<(MROWS / 16) * (Cc / 16) / 8, 256, 0, stream>>>(
      ybf, W_proj ? Wpbf : Wpbf, out);
}
